// Pointnet2Backbone_22342419873990
// MI455X (gfx1250) — matrix operand order repaired and checked by reading, not compiled, not run on hardware
//
#include <hip/hip_runtime.h>
#include <hip/hip_bf16.h>

typedef _Float16 half_t;
typedef __attribute__((ext_vector_type(16))) _Float16 v16h;
typedef __attribute__((ext_vector_type(8)))  float    v8f;
typedef __attribute__((ext_vector_type(2)))  _Float16 h2;

static inline int cdiv(int a, int b) { return (a + b - 1) / b; }

// ---------------------------------------------------------------------------
// Weight prep: W fp32 [cin,cout] -> Wt fp16 transposed+padded [cout, Kp]
// ---------------------------------------------------------------------------
__global__ void prep_weight_kernel(const float* __restrict__ W, half_t* __restrict__ Wt,
                                   int cin, int cout, int Kp) {
  int i = blockIdx.x * blockDim.x + threadIdx.x;
  if (i >= cout * Kp) return;
  int n = i / Kp, k = i % Kp;
  float v = (k < cin) ? W[(size_t)k * cout + n] : 0.0f;
  Wt[i] = (half_t)v;
}

// ---------------------------------------------------------------------------
// Farthest point sampling: one block per batch; dists kept in global scratch.
// ---------------------------------------------------------------------------
__global__ void fps_kernel(const float* __restrict__ xyz, int N, int npoint,
                           int* __restrict__ inds, float* __restrict__ dists) {
  int b = blockIdx.x;
  const float* p = xyz + (size_t)b * N * 3;
  float* dd = dists + (size_t)b * N;
  __shared__ float smax[256];
  __shared__ int   simx[256];
  __shared__ int   slast;
  int tid = threadIdx.x;
  for (int i = tid; i < N; i += 256) dd[i] = 1e10f;
  if (tid == 0) { inds[b * npoint] = 0; slast = 0; }
  __syncthreads();
  for (int it = 1; it < npoint; ++it) {
    int last = slast;
    float lx = p[last * 3], ly = p[last * 3 + 1], lz = p[last * 3 + 2];
    float bm = -1.0f; int bi = 0;
    for (int i = tid; i < N; i += 256) {
      float dx = p[i * 3] - lx, dy = p[i * 3 + 1] - ly, dz = p[i * 3 + 2] - lz;
      float d = dx * dx + dy * dy + dz * dz;
      float dn = dd[i]; dn = d < dn ? d : dn; dd[i] = dn;
      if (dn > bm) { bm = dn; bi = i; }
    }
    smax[tid] = bm; simx[tid] = bi;
    __syncthreads();
    for (int s = 128; s > 0; s >>= 1) {
      if (tid < s) {
        float o = smax[tid + s];
        if (o > smax[tid] || (o == smax[tid] && simx[tid + s] < simx[tid])) {
          smax[tid] = o; simx[tid] = simx[tid + s];
        }
      }
      __syncthreads();
    }
    if (tid == 0) { slast = simx[0]; inds[b * npoint + it] = simx[0]; }
    __syncthreads();
  }
}

// ---------------------------------------------------------------------------
// Gather sampled centroids.
// ---------------------------------------------------------------------------
__global__ void gather_xyz_kernel(const float* __restrict__ xyz, const int* __restrict__ inds,
                                  float* __restrict__ out, int Nin, int np, int B) {
  int i = blockIdx.x * blockDim.x + threadIdx.x;
  if (i >= B * np) return;
  int b = i / np;
  int j = inds[i];
  const float* s = xyz + ((size_t)b * Nin + j) * 3;
  out[i * 3] = s[0]; out[i * 3 + 1] = s[1]; out[i * 3 + 2] = s[2];
}

// ---------------------------------------------------------------------------
// Ball query: first `ns` points (by index) within radius; pad with first hit.
// Matches reference top_k(-index) semantics.
// ---------------------------------------------------------------------------
__global__ void ball_query_kernel(const float* __restrict__ xyz, const float* __restrict__ nxyz,
                                  int* __restrict__ idx, int N, int np, int ns, float r2, int B) {
  int g = blockIdx.x * blockDim.x + threadIdx.x;
  if (g >= B * np) return;
  int b = g / np;
  const float* p = xyz + (size_t)b * N * 3;
  float cx = nxyz[g * 3], cy = nxyz[g * 3 + 1], cz = nxyz[g * 3 + 2];
  int* o = idx + (size_t)g * ns;
  int cnt = 0, first = 0;
  for (int j = 0; j < N && cnt < ns; ++j) {
    float dx = p[j * 3] - cx, dy = p[j * 3 + 1] - cy, dz = p[j * 3 + 2] - cz;
    float d = dx * dx + dy * dy + dz * dz;
    if (d < r2) { if (cnt == 0) first = j; o[cnt++] = j; }
  }
  for (int k = cnt; k < ns; ++k) o[k] = first;
}

// ---------------------------------------------------------------------------
// Grouping: A[(b,s,k), c] = c<3 ? (xyz[idx]-center)/r : feats[idx, c-3]; pad 0.
// Writes the fp16 GEMM A-matrix directly (K padded to Kp, multiple of 32).
// ---------------------------------------------------------------------------
__global__ void group_kernel(const float* __restrict__ xyz, const float* __restrict__ nxyz,
                             const float* __restrict__ feats, const int* __restrict__ idx,
                             half_t* __restrict__ A, int Nin, int np, int ns,
                             int Cf, int Kp, float invr, int B) {
  int i = blockIdx.x * blockDim.x + threadIdx.x;
  int total = B * np * ns * Kp;
  if (i >= total) return;
  int c = i % Kp;
  int row = i / Kp;
  int s = (row / ns) % np;
  int b = row / (ns * np);
  int j = idx[row];
  float v = 0.0f;
  if (c < 3) {
    v = (xyz[((size_t)b * Nin + j) * 3 + c] - nxyz[((size_t)b * np + s) * 3 + c]) * invr;
  } else if (c < 3 + Cf) {
    v = feats[((size_t)b * Nin + j) * Cf + (c - 3)];
  }
  A[i] = (half_t)v;
}

// ---------------------------------------------------------------------------
// WMMA GEMM: Y[M,N] fp32 = A[M,KP] fp16 x Bt[N,KP]^T fp16.  One wave per
// 16x16 tile; KP is a compile-time constant so the whole K loop unrolls into
// a straight chain of v_wmma_f32_16x16x32_f16 fed by global_load_b128 clauses.
// Fragment layouts per CDNA5 ISA 7.12.2 (16-bit A 16x32, B 32x16, f32 C/D).
// ---------------------------------------------------------------------------
template <int KP>
__global__ void __launch_bounds__(256) gemm_wmma_kernel(
    const half_t* __restrict__ A, const half_t* __restrict__ Bt,
    float* __restrict__ Y, int M, int N) {
  int lane = threadIdx.x & 31;
  int wave = threadIdx.x >> 5;
  int tilesN = N >> 4;
  int totalTiles = (M >> 4) * tilesN;
  int tile = blockIdx.x * (blockDim.x >> 5) + wave;
  if (tile >= totalTiles) return;   // wave-uniform exit: EXEC stays all-1s
  int tm = (tile / tilesN) << 4;
  int tn = (tile % tilesN) << 4;
  int hl = lane >> 4;               // lane half (0/1)
  int l16 = lane & 15;
  const half_t* arow = A  + (size_t)(tm + l16) * KP;
  const half_t* brow = Bt + (size_t)(tn + l16) * KP;
  v8f c = {};
#pragma unroll
  for (int k0 = 0; k0 < KP; k0 += 32) {
    v16h a, b;
#pragma unroll
    for (int j = 0; j < 8; ++j) {
      // A: VGPR j holds K = (j%4)*2 + (j/4)*16 + half*8 + {0,1}
      int ka = ((j & 3) << 1) + ((j >> 2) << 4) + (hl << 3);
      h2 pa = *(const h2*)(arow + k0 + ka);
      a[2 * j] = pa[0]; a[2 * j + 1] = pa[1];
      // B: VGPR j holds K = (j%4)*2 + (j/4)*16 + half*8 + {0,1} (same as A)
      int kb = ((j & 3) << 1) + ((j >> 2) << 4) + (hl << 3);
      h2 pb = *(const h2*)(brow + k0 + kb);
      b[2 * j] = pb[0]; b[2 * j + 1] = pb[1];
    }
    c = __builtin_amdgcn_wmma_f32_16x16x32_f16(false, a, false, b, (short)0, c, false, false);
  }
#pragma unroll
  for (int r = 0; r < 8; ++r)
    Y[(size_t)(tm + r + hl * 8) * N + tn + l16] = c[r];
}

// ---------------------------------------------------------------------------
// BatchNorm pass 1: per-channel mean and rsqrt(var+eps). One block per channel.
// ---------------------------------------------------------------------------
__global__ void bn_stats_kernel(const float* __restrict__ Y, float* __restrict__ mean,
                                float* __restrict__ rsig, int M, int N) {
  int c = blockIdx.x;
  __shared__ float s1[256], s2[256];
  int tid = threadIdx.x;
  float a = 0.0f, b = 0.0f;
  for (int m = tid; m < M; m += 256) {
    float v = Y[(size_t)m * N + c];
    a += v; b += v * v;
  }
  s1[tid] = a; s2[tid] = b;
  __syncthreads();
  for (int s = 128; s > 0; s >>= 1) {
    if (tid < s) { s1[tid] += s1[tid + s]; s2[tid] += s2[tid + s]; }
    __syncthreads();
  }
  if (tid == 0) {
    float mu = s1[0] / (float)M;
    float var = s2[0] / (float)M - mu * mu;
    mean[c] = mu;
    rsig[c] = rsqrtf(var + 1e-5f);
  }
}

// ---------------------------------------------------------------------------
// BatchNorm pass 2 + ReLU; optionally emit next layer's fp16 A (Kp_next == N).
// ---------------------------------------------------------------------------
__global__ void bn_relu_kernel(float* __restrict__ Y, const float* __restrict__ mean,
                               const float* __restrict__ rsig, const float* __restrict__ gamma,
                               const float* __restrict__ beta, half_t* __restrict__ Anext,
                               int M, int N) {
  int i = blockIdx.x * blockDim.x + threadIdx.x;
  if (i >= M * N) return;
  int c = i % N;
  float v = (Y[i] - mean[c]) * rsig[c] * gamma[c] + beta[c];
  v = fmaxf(v, 0.0f);
  Y[i] = v;
  if (Anext) Anext[i] = (half_t)v;
}

// ---------------------------------------------------------------------------
// Max-pool over the nsample dimension: F[b,s,c] = max_k Y[(b,s,k),c].
// ---------------------------------------------------------------------------
__global__ void maxpool_kernel(const float* __restrict__ Y, float* __restrict__ F,
                               int np, int ns, int C, int B) {
  int i = blockIdx.x * blockDim.x + threadIdx.x;
  if (i >= B * np * C) return;
  int c = i % C;
  int srow = i / C;
  const float* base = Y + (size_t)srow * ns * C + c;
  float m = -1e30f;
  for (int k = 0; k < ns; ++k) m = fmaxf(m, base[(size_t)k * C]);
  F[i] = m;
}

// ---------------------------------------------------------------------------
// 3-NN (squared distances, like the reference) + interpolation weights.
// ---------------------------------------------------------------------------
__global__ void three_nn_kernel(const float* __restrict__ unknown, const float* __restrict__ known,
                                int n, int m, int* __restrict__ idx3, float* __restrict__ w3, int B) {
  int g = blockIdx.x * blockDim.x + threadIdx.x;
  if (g >= B * n) return;
  int b = g / n;
  float ux = unknown[g * 3], uy = unknown[g * 3 + 1], uz = unknown[g * 3 + 2];
  const float* kp = known + (size_t)b * m * 3;
  float b0 = 1e30f, b1 = 1e30f, b2 = 1e30f;
  int i0 = 0, i1 = 0, i2 = 0;
  for (int j = 0; j < m; ++j) {
    float dx = kp[j * 3] - ux, dy = kp[j * 3 + 1] - uy, dz = kp[j * 3 + 2] - uz;
    float d = dx * dx + dy * dy + dz * dz;
    if (d < b0)      { b2 = b1; i2 = i1; b1 = b0; i1 = i0; b0 = d; i0 = j; }
    else if (d < b1) { b2 = b1; i2 = i1; b1 = d;  i1 = j; }
    else if (d < b2) { b2 = d;  i2 = j; }
  }
  float w0 = 1.0f / (b0 + 1e-8f), w1 = 1.0f / (b1 + 1e-8f), w2 = 1.0f / (b2 + 1e-8f);
  float s = w0 + w1 + w2;
  idx3[g * 3] = i0; idx3[g * 3 + 1] = i1; idx3[g * 3 + 2] = i2;
  w3[g * 3] = w0 / s; w3[g * 3 + 1] = w1 / s; w3[g * 3 + 2] = w2 / s;
}

// ---------------------------------------------------------------------------
// Interpolate kn_feats with 3-NN weights, concat with skip feats -> fp16 A.
// ---------------------------------------------------------------------------
__global__ void interp_concat_kernel(const float* __restrict__ kn_feats,
                                     const float* __restrict__ unk_feats,
                                     const int* __restrict__ idx3, const float* __restrict__ w3,
                                     half_t* __restrict__ A, int n, int m, int C, int Cs,
                                     int Kp, int B) {
  int i = blockIdx.x * blockDim.x + threadIdx.x;
  int total = B * n * Kp;
  if (i >= total) return;
  int c = i % Kp;
  int row = i / Kp;
  int b = row / n;
  float v;
  if (c < C) {
    const int* id = idx3 + row * 3;
    const float* w = w3 + row * 3;
    const float* base = kn_feats + (size_t)b * m * C + c;
    v = w[0] * base[(size_t)id[0] * C] + w[1] * base[(size_t)id[1] * C] + w[2] * base[(size_t)id[2] * C];
  } else {
    v = unk_feats[(size_t)row * Cs + (c - C)];
  }
  A[i] = (half_t)v;
}

// ---------------------------------------------------------------------------
// Final output: [xyz2 flat | transpose(f2n,(0,2,1)) flat]
// ---------------------------------------------------------------------------
__global__ void out_kernel(const float* __restrict__ xyz2, const float* __restrict__ f2n,
                           float* __restrict__ out, int B, int n, int C) {
  int i = blockIdx.x * blockDim.x + threadIdx.x;
  int nxyz = B * n * 3;
  int nf = B * C * n;
  if (i < nxyz) {
    out[i] = xyz2[i];
  } else if (i < nxyz + nf) {
    int j = i - nxyz;
    int b = j / (C * n);
    int r = j % (C * n);
    int c = r / n;
    int p = r % n;
    out[i] = f2n[((size_t)(b * n + p)) * C + c];
  }
}

// ---------------------------------------------------------------------------
// Host orchestration
// ---------------------------------------------------------------------------
extern "C" void kernel_launch(void* const* d_in, const int* in_sizes, int n_in,
                              void* d_out, int out_size, void* d_ws, size_t ws_size,
                              hipStream_t stream) {
  (void)in_sizes; (void)n_in; (void)out_size; (void)ws_size;
  const float* pc = (const float*)d_in[0];   // [8,16384,3]

  struct LD { int cin, cout, kp; };
  const LD L[16] = {
    {3, 64, 32},   {64, 64, 64},   {64, 128, 64},     // sa1
    {131, 128, 160},{128, 128, 128},{128, 256, 128},  // sa2
    {259, 128, 288},{128, 128, 128},{128, 256, 128},  // sa3
    {259, 128, 288},{128, 128, 128},{128, 256, 128},  // sa4
    {512, 256, 512},{256, 256, 256},                  // fp1
    {512, 256, 512},{256, 288, 256},                  // fp2
  };
  const float *W[16], *G[16], *Bp[16];
  int pi = 1;
  for (int l = 0; l < 16; ++l) {
    W[l]  = (const float*)d_in[pi];
    G[l]  = (const float*)d_in[pi + 1];
    Bp[l] = (const float*)d_in[pi + 2];
    pi += 3;
  }

  // Bump-allocate scratch (~175 MB peak).
  char* ws = (char*)d_ws;
  size_t off = 0;
  auto alloc = [&](size_t bytes) -> void* {
    void* r = ws + off;
    off = (off + bytes + 255) & ~(size_t)255;
    return r;
  };
  half_t* Wt[16];
  for (int l = 0; l < 16; ++l) Wt[l] = (half_t*)alloc((size_t)L[l].cout * L[l].kp * sizeof(half_t));
  float* dists = (float*)alloc((size_t)8 * 16384 * sizeof(float));
  int*   inds  = (int*)  alloc((size_t)8 * 512 * sizeof(int));
  int*   idxb  = (int*)  alloc((size_t)8 * 512 * 64 * sizeof(int));
  const int nps[4] = {512, 256, 128, 64};
  float* xyzs[4];
  for (int s = 0; s < 4; ++s) xyzs[s] = (float*)alloc((size_t)8 * nps[s] * 3 * sizeof(float));
  float* f1  = (float*)alloc((size_t)8 * 512 * 128 * sizeof(float));
  float* f2  = (float*)alloc((size_t)8 * 256 * 256 * sizeof(float));
  float* f3  = (float*)alloc((size_t)8 * 128 * 256 * sizeof(float));
  float* f4  = (float*)alloc((size_t)8 * 64  * 256 * sizeof(float));
  float* f3n = (float*)alloc((size_t)8 * 128 * 256 * sizeof(float));
  int*   idx3 = (int*)  alloc((size_t)8 * 256 * 3 * sizeof(int));
  float* w3   = (float*)alloc((size_t)8 * 256 * 3 * sizeof(float));
  float* meanb = (float*)alloc(512 * sizeof(float));
  float* rsigb = (float*)alloc(512 * sizeof(float));
  half_t* Abuf = (half_t*)alloc((size_t)262144 * 64 * sizeof(half_t));   // 32 MB
  float*  Ybuf = (float*) alloc((size_t)262144 * 128 * sizeof(float));   // 128 MB

  // Prepare all weights (transpose + pad + fp16).
  for (int l = 0; l < 16; ++l) {
    int tot = L[l].cout * L[l].kp;
    prep_weight_kernel<<<cdiv(tot, 256), 256, 0, stream>>>(W[l], Wt[l], L[l].cin, L[l].cout, L[l].kp);
  }

  // Fully-unrolled WMMA GEMM dispatch over the compile-time K set.
  auto launch_gemm = [&](const half_t* A, const half_t* Bt, float* Y, int M, int N, int Kp) {
    int tiles = (M / 16) * (N / 16);
    int blocks = cdiv(tiles, 8);
    switch (Kp) {
      case 32:  gemm_wmma_kernel<32 ><<<blocks, 256, 0, stream>>>(A, Bt, Y, M, N); break;
      case 64:  gemm_wmma_kernel<64 ><<<blocks, 256, 0, stream>>>(A, Bt, Y, M, N); break;
      case 128: gemm_wmma_kernel<128><<<blocks, 256, 0, stream>>>(A, Bt, Y, M, N); break;
      case 160: gemm_wmma_kernel<160><<<blocks, 256, 0, stream>>>(A, Bt, Y, M, N); break;
      case 256: gemm_wmma_kernel<256><<<blocks, 256, 0, stream>>>(A, Bt, Y, M, N); break;
      case 288: gemm_wmma_kernel<288><<<blocks, 256, 0, stream>>>(A, Bt, Y, M, N); break;
      case 512: gemm_wmma_kernel<512><<<blocks, 256, 0, stream>>>(A, Bt, Y, M, N); break;
      default:  break; // all layer shapes are covered above
    }
  };

  auto run_layer = [&](int l, int M, bool writeNext) {
    int N = L[l].cout, Kp = L[l].kp;
    launch_gemm(Abuf, Wt[l], Ybuf, M, N, Kp);
    bn_stats_kernel<<<N, 256, 0, stream>>>(Ybuf, meanb, rsigb, M, N);
    bn_relu_kernel<<<cdiv(M * N, 256), 256, 0, stream>>>(
        Ybuf, meanb, rsigb, G[l], Bp[l], writeNext ? Abuf : (half_t*)nullptr, M, N);
  };

  // -------- Set-abstraction stages --------
  struct SA { int Nin, np, ns; float r; int l0, Cf; };
  const SA sas[4] = {
    {16384, 512, 64, 0.2f, 0, 0},
    {512,   256, 32, 0.4f, 3, 128},
    {256,   128, 16, 0.8f, 6, 256},
    {128,   64,  16, 1.2f, 9, 256},
  };
  float* fouts[4] = {f1, f2, f3, f4};
  const float* xin = pc;
  const float* fin = nullptr;
  for (int s = 0; s < 4; ++s) {
    const SA& c = sas[s];
    fps_kernel<<<8, 256, 0, stream>>>(xin, c.Nin, c.np, inds, dists);
    int q = 8 * c.np;
    gather_xyz_kernel<<<cdiv(q, 256), 256, 0, stream>>>(xin, inds, xyzs[s], c.Nin, c.np, 8);
    ball_query_kernel<<<cdiv(q, 256), 256, 0, stream>>>(xin, xyzs[s], idxb, c.Nin, c.np, c.ns,
                                                        c.r * c.r, 8);
    int M = 8 * c.np * c.ns;
    int Kp0 = L[c.l0].kp;
    group_kernel<<<cdiv(M * Kp0, 256), 256, 0, stream>>>(xin, xyzs[s], fin, idxb, Abuf,
                                                         c.Nin, c.np, c.ns, c.Cf, Kp0,
                                                         1.0f / c.r, 8);
    run_layer(c.l0, M, true);
    run_layer(c.l0 + 1, M, true);
    run_layer(c.l0 + 2, M, false);
    int Cl = L[c.l0 + 2].cout;
    maxpool_kernel<<<cdiv(8 * c.np * Cl, 256), 256, 0, stream>>>(Ybuf, fouts[s], c.np, c.ns, Cl, 8);
    xin = xyzs[s];
    fin = fouts[s];
  }

  // -------- Feature propagation 1: (xyz3, xyz4, f3, f4) -> f3n --------
  {
    int n = 128, m = 64, C = 256, Cs = 256, Kp = 512, M = 8 * n;
    three_nn_kernel<<<cdiv(8 * n, 256), 256, 0, stream>>>(xyzs[2], xyzs[3], n, m, idx3, w3, 8);
    interp_concat_kernel<<<cdiv(M * Kp, 256), 256, 0, stream>>>(f4, f3, idx3, w3, Abuf,
                                                                n, m, C, Cs, Kp, 8);
    run_layer(12, M, true);
    run_layer(13, M, false);
    hipMemcpyAsync(f3n, Ybuf, (size_t)M * 256 * sizeof(float), hipMemcpyDeviceToDevice, stream);
  }

  // -------- Feature propagation 2: (xyz2, xyz3, f2, f3n) -> f2n --------
  {
    int n = 256, m = 128, C = 256, Cs = 256, Kp = 512, M = 8 * n;
    three_nn_kernel<<<cdiv(8 * n, 256), 256, 0, stream>>>(xyzs[1], xyzs[2], n, m, idx3, w3, 8);
    interp_concat_kernel<<<cdiv(M * Kp, 256), 256, 0, stream>>>(f3n, f2, idx3, w3, Abuf,
                                                                n, m, C, Cs, Kp, 8);
    run_layer(14, M, true);
    run_layer(15, M, false);   // f2n now lives in Ybuf [2048, 288]
  }

  // -------- Output: xyz2 then transpose(f2n, (0,2,1)) --------
  {
    int B = 8, n = 256, C = 288;
    int tot = B * n * 3 + B * C * n;
    out_kernel<<<cdiv(tot, 256), 256, 0, stream>>>(xyzs[1], Ybuf, (float*)d_out, B, n, C);
  }
}
